// TheExpertAbundance_15479062135478
// MI455X (gfx1250) — compile-verified, matrix operand order checked by reading, not run on hardware
//
#include <hip/hip_runtime.h>

// ---------------------------------------------------------------------------
// Types for CDNA5 WMMA (wave32)
// ---------------------------------------------------------------------------
typedef __bf16 bf16_t;
typedef __attribute__((ext_vector_type(16))) __bf16 v16bf;
typedef __attribute__((ext_vector_type(8)))  float  v8f;

static __device__ __forceinline__ bf16_t f2bf(float f) {
    unsigned u = __float_as_uint(f);
    u += 0x7FFFu + ((u >> 16) & 1u);           // round to nearest even
    unsigned short h = (unsigned short)(u >> 16);
    return __builtin_bit_cast(bf16_t, h);
}

// Load a 16-element bf16 fragment as two 16-byte chunks (matches the
// documented 16-bit A/B VGPR layout: K 0..7 / 16..23 for lanes 0-15,
// K 8..15 / 24..31 for lanes 16-31 -> caller passes khalf-adjusted ptrs).
static __device__ __forceinline__ v16bf load_frag(const bf16_t* p0, const bf16_t* p1) {
    union { v16bf v; uint4 u[2]; } f;
    f.u[0] = *reinterpret_cast<const uint4*>(p0);
    f.u[1] = *reinterpret_cast<const uint4*>(p1);
    return f.v;
}

static __device__ __forceinline__ v8f wmma_bf16(v16bf a, v16bf b, v8f c) {
    return __builtin_amdgcn_wmma_f32_16x16x32_bf16(
        /*neg_a=*/false, a, /*neg_b=*/false, b,
        /*c_mod=*/(short)0, c, /*reuse_a=*/false, /*reuse_b=*/false);
}

// ---------------------------------------------------------------------------
// Problem constants
// ---------------------------------------------------------------------------
#define BB   2
#define TT   2048
#define EE   2048
#define HH   16
#define DD   128
#define NN   (3 * EE)        // 6144
#define MM   (BB * TT)       // 4096

// ---------------------------------------------------------------------------
// Kernel 1: RMS-norm rows of x -> bf16, and copy x (fp32) into d_out half 0
// ---------------------------------------------------------------------------
__global__ __launch_bounds__(256) void rmsnorm_x_kernel(
    const float* __restrict__ x, bf16_t* __restrict__ xn,
    float* __restrict__ xout)
{
    __shared__ float red[256];
    const int row = blockIdx.x;
    const float* xr = x + (size_t)row * EE;
    float ssq = 0.f;
    for (int i = threadIdx.x; i < EE; i += 256) {
        float v = xr[i];
        ssq += v * v;
    }
    red[threadIdx.x] = ssq;
    __syncthreads();
    for (int s = 128; s > 0; s >>= 1) {
        if (threadIdx.x < s) red[threadIdx.x] += red[threadIdx.x + s];
        __syncthreads();
    }
    const float r = rsqrtf(red[0] * (1.0f / EE) + 1.1920929e-7f);
    for (int i = threadIdx.x; i < EE; i += 256) {
        float v = xr[i];
        xout[(size_t)row * EE + i] = v;                  // passthrough output
        xn[(size_t)row * EE + i]  = f2bf(v * r);
    }
}

// ---------------------------------------------------------------------------
// Kernel 2: convert w_qkv fp32 -> bf16 (grid-stride)
// ---------------------------------------------------------------------------
__global__ __launch_bounds__(256) void convert_w_kernel(
    const float* __restrict__ w, bf16_t* __restrict__ wb, int n)
{
    for (int i = blockIdx.x * 256 + threadIdx.x; i < n; i += gridDim.x * 256)
        wb[i] = f2bf(w[i]);
}

// ---------------------------------------------------------------------------
// Kernel 3: QKV GEMM  C[M,N] = A[M,K] * W[N,K]^T   (bf16 in, fp32 out)
//   Block tile 128x256, 8 waves (2Mx4N), wave tile 64x64, K-step 32.
//   16 WMMAs per 8 fragment loads per wave per K-step.
// ---------------------------------------------------------------------------
#define LDSS 40   // lds row stride in bf16 elems (80B, keeps 16B alignment)

__global__ __launch_bounds__(256) void qkv_gemm_kernel(
    const bf16_t* __restrict__ A, const bf16_t* __restrict__ Bw,
    float* __restrict__ C)
{
    __shared__ bf16_t As[128][LDSS];
    __shared__ bf16_t Bs[256][LDSS];

    const int tid   = threadIdx.x;
    const int lane  = tid & 31;
    const int wave  = tid >> 5;        // 0..7
    const int waveM = wave >> 2;       // 0..1  (64 rows each)
    const int waveN = wave & 3;        // 0..3  (64 cols each)
    const int lm    = lane & 15;
    const int hl    = lane >> 4;
    const int khalf = hl * 8;

    const int bm = (int)blockIdx.x * 128;
    const int bn = (int)blockIdx.y * 256;

    // staging map: A tile 128x32 -> 1 chunk/thread; B tile 256x32 -> 2 chunks
    const int srow  = tid >> 1;        // 0..127
    const int shalf = (tid & 1) * 16;  // 0 or 16

    const bf16_t* aptr  = A  + (size_t)(bm + srow) * EE + shalf;
    const bf16_t* bptr0 = Bw + (size_t)(bn + srow) * EE + shalf;
    const bf16_t* bptr1 = bptr0 + (size_t)128 * EE;

    v8f acc[4][4];
    for (int i = 0; i < 4; ++i)
        for (int j = 0; j < 4; ++j)
            for (int e = 0; e < 8; ++e) acc[i][j][e] = 0.f;

    for (int k0 = 0; k0 < EE; k0 += 32) {
        __syncthreads();
        *reinterpret_cast<uint4*>(&As[srow][shalf]) =
            *reinterpret_cast<const uint4*>(aptr + k0);
        *reinterpret_cast<uint4*>(&Bs[srow][shalf]) =
            *reinterpret_cast<const uint4*>(bptr0 + k0);
        *reinterpret_cast<uint4*>(&Bs[128 + srow][shalf]) =
            *reinterpret_cast<const uint4*>(bptr1 + k0);
        if (k0 + 32 < EE) {           // warm L2 for next K-step (global_prefetch_b8)
            __builtin_prefetch(aptr  + k0 + 32, 0, 1);
            __builtin_prefetch(bptr0 + k0 + 32, 0, 1);
            __builtin_prefetch(bptr1 + k0 + 32, 0, 1);
        }
        __syncthreads();

        v16bf af[4], bfrag[4];
        for (int i = 0; i < 4; ++i) {
            const int row = waveM * 64 + i * 16 + lm;
            af[i] = load_frag(&As[row][khalf], &As[row][16 + khalf]);
        }
        for (int j = 0; j < 4; ++j) {
            const int row = waveN * 64 + j * 16 + lm;
            bfrag[j] = load_frag(&Bs[row][khalf], &Bs[row][16 + khalf]);
        }
        for (int i = 0; i < 4; ++i)
            for (int j = 0; j < 4; ++j)
                acc[i][j] = wmma_bf16(af[i], bfrag[j], acc[i][j]);
    }

    for (int i = 0; i < 4; ++i)
        for (int j = 0; j < 4; ++j)
            for (int e = 0; e < 8; ++e) {
                const int row = bm + waveM * 64 + i * 16 + e + 8 * hl;
                const int col = bn + waveN * 64 + j * 16 + lm;
                C[(size_t)row * NN + col] = acc[i][j][e];
            }
}

// ---------------------------------------------------------------------------
// Kernel 4: RoPE + per-head RMS-norm for Q/K; layout for attention:
//   qh, kh : [B,H,T,D] bf16      vt : [B,H,D,T] bf16 (transposed V)
//   One block = one (b,t,h), 128 threads (one per head element).
// ---------------------------------------------------------------------------
__global__ __launch_bounds__(128) void rope_rms_kernel(
    const float* __restrict__ qkv, const float* __restrict__ cosp,
    const float* __restrict__ sinp, bf16_t* __restrict__ qh,
    bf16_t* __restrict__ kh, bf16_t* __restrict__ vt)
{
    const int idx = blockIdx.x;
    const int h   = idx & (HH - 1);
    const int bt  = idx >> 4;            // b*T + t
    const int t   = bt & (TT - 1);
    const int b   = bt / TT;
    const int tid = threadIdx.x;

    const float* row = qkv + (size_t)bt * NN + h * (3 * DD);

    __shared__ float qs[DD];
    __shared__ float ks[DD];
    __shared__ float red[DD];

    if (tid < 64) {
        const float x1 = row[tid], x2 = row[tid + 64];
        const float c = cosp[t * 64 + tid], s = sinp[t * 64 + tid];
        qs[tid]      =  x1 * c + x2 * s;
        qs[tid + 64] = -x1 * s + x2 * c;
    } else {
        const int i = tid - 64;
        const float x1 = row[DD + i], x2 = row[DD + i + 64];
        const float c = cosp[t * 64 + i], s = sinp[t * 64 + i];
        ks[i]      =  x1 * c + x2 * s;
        ks[i + 64] = -x1 * s + x2 * c;
    }
    __syncthreads();

    const float qv = qs[tid];
    const float kv = ks[tid];

    red[tid] = qv * qv;
    __syncthreads();
    for (int s = 64; s > 0; s >>= 1) {
        if (tid < s) red[tid] += red[tid + s];
        __syncthreads();
    }
    const float rq = rsqrtf(red[0] * (1.0f / DD) + 1.1920929e-7f);
    __syncthreads();

    red[tid] = kv * kv;
    __syncthreads();
    for (int s = 64; s > 0; s >>= 1) {
        if (tid < s) red[tid] += red[tid + s];
        __syncthreads();
    }
    const float rk = rsqrtf(red[0] * (1.0f / DD) + 1.1920929e-7f);

    const size_t head = (size_t)b * HH + h;
    qh[(head * TT + t) * DD + tid] = f2bf(qv * rq);
    kh[(head * TT + t) * DD + tid] = f2bf(kv * rk);
    vt[(head * DD + tid) * TT + t] = f2bf(row[2 * DD + tid]);
}

// ---------------------------------------------------------------------------
// Kernel 5: causal flash attention.
//   One block = (b, h, 64-query tile), 4 waves x 16 query rows.
//   Per 32-key block: 8 WMMAs for S = Q K^T, online softmax (shfl reductions
//   inside 16-lane halves, matching C layout), P -> bf16 via LDS (layout
//   transpose C->A), 8 WMMAs for O += P V (V pre-transposed [D,T]).
// ---------------------------------------------------------------------------
__global__ __launch_bounds__(128) void flash_attn_kernel(
    const bf16_t* __restrict__ qh, const bf16_t* __restrict__ kh,
    const bf16_t* __restrict__ vt, float* __restrict__ y)
{
    __shared__ bf16_t Plds[4][16][LDSS];

    const int tid   = threadIdx.x;
    const int lane  = tid & 31;
    const int wave  = tid >> 5;      // 0..3
    const int lm    = lane & 15;
    const int hl    = lane >> 4;
    const int khalf = hl * 8;

    const int blk = blockIdx.x;
    const int qb  = blk & (TT / 64 - 1);
    const int bh  = blk / (TT / 64);       // b*H + h
    const int h   = bh & (HH - 1);
    const int b   = bh / HH;

    const int q0    = qb * 64;
    const int qrow0 = q0 + wave * 16;

    const bf16_t* qbase = qh + (size_t)bh * TT * DD;
    const bf16_t* kbase = kh + (size_t)bh * TT * DD;
    const bf16_t* vbase = vt + (size_t)bh * DD * TT;

    // Q fragments: 4 K-steps covering D=128
    v16bf qf[4];
    for (int s = 0; s < 4; ++s) {
        const bf16_t* p = qbase + (size_t)(qrow0 + lm) * DD + s * 32 + khalf;
        qf[s] = load_frag(p, p + 16);
    }

    v8f o[8];
    for (int dt = 0; dt < 8; ++dt)
        for (int e = 0; e < 8; ++e) o[dt][e] = 0.f;

    float mrow[8], lrow[8];
    for (int e = 0; e < 8; ++e) { mrow[e] = -__builtin_inff(); lrow[e] = 0.f; }

    const float scale = 0.08838834764831845f;   // 1/sqrt(128)
    const int   jend  = q0 + 64;

    for (int j = 0; j < jend; j += 32) {
        // ---- S = Q K^T (16 rows x 32 keys) ----
        v8f s0, s1;
        for (int e = 0; e < 8; ++e) { s0[e] = 0.f; s1[e] = 0.f; }
        for (int s = 0; s < 4; ++s) {
            const bf16_t* kp0 = kbase + (size_t)(j + lm) * DD + s * 32 + khalf;
            const bf16_t* kp1 = kbase + (size_t)(j + 16 + lm) * DD + s * 32 + khalf;
            v16bf kf0 = load_frag(kp0, kp0 + 16);
            v16bf kf1 = load_frag(kp1, kp1 + 16);
            s0 = wmma_bf16(qf[s], kf0, s0);
            s1 = wmma_bf16(qf[s], kf1, s1);
        }

        // ---- causal mask + online softmax ----
        float rs[8];
        for (int e = 0; e < 8; ++e) {
            const int row = qrow0 + e + 8 * hl;
            const int c0  = j + lm;
            const int c1  = j + 16 + lm;
            float a  = (c0 <= row) ? s0[e] * scale : -__builtin_inff();
            float bq = (c1 <= row) ? s1[e] * scale : -__builtin_inff();
            float mx = fmaxf(a, bq);
            for (int d = 1; d < 16; d <<= 1) mx = fmaxf(mx, __shfl_xor(mx, d, 32));
            const float mnew = fmaxf(mrow[e], mx);
            rs[e] = __expf(mrow[e] - mnew);
            const float p0 = __expf(a - mnew);
            const float p1 = __expf(bq - mnew);
            float psum = p0 + p1;
            for (int d = 1; d < 16; d <<= 1) psum += __shfl_xor(psum, d, 32);
            lrow[e] = lrow[e] * rs[e] + psum;
            mrow[e] = mnew;
            s0[e] = p0;
            s1[e] = p1;
        }

        // ---- P (C layout) -> bf16 LDS -> A layout fragment ----
        for (int e = 0; e < 8; ++e) {
            const int r = e + 8 * hl;
            Plds[wave][r][lm]      = f2bf(s0[e]);
            Plds[wave][r][16 + lm] = f2bf(s1[e]);
        }
        __syncthreads();
        const bf16_t* pp = &Plds[wave][lm][khalf];
        v16bf pf = load_frag(pp, pp + 16);

        // ---- O = O*rs + P V ----
        for (int dt = 0; dt < 8; ++dt) {
            const bf16_t* vp = vbase + (size_t)(dt * 16 + lm) * TT + j + khalf;
            v16bf vf = load_frag(vp, vp + 16);
            v8f oo = o[dt];
            for (int e = 0; e < 8; ++e) oo[e] *= rs[e];
            o[dt] = wmma_bf16(pf, vf, oo);
        }
        __syncthreads();
    }

    // ---- finalize: y[b, t, h, d] = O / l ----
    for (int dt = 0; dt < 8; ++dt)
        for (int e = 0; e < 8; ++e) {
            const int trow = qrow0 + e + 8 * hl;
            const int d    = dt * 16 + lm;
            y[(((size_t)b * TT + trow) * HH + h) * DD + d] = o[dt][e] / lrow[e];
        }
}

// ---------------------------------------------------------------------------
// Launcher
// ---------------------------------------------------------------------------
extern "C" void kernel_launch(void* const* d_in, const int* in_sizes, int n_in,
                              void* d_out, int out_size, void* d_ws, size_t ws_size,
                              hipStream_t stream)
{
    (void)in_sizes; (void)n_in; (void)out_size; (void)ws_size;

    const float* x    = (const float*)d_in[0];
    const float* cosp = (const float*)d_in[1];
    const float* sinp = (const float*)d_in[2];
    const float* w    = (const float*)d_in[3];

    char* ws = (char*)d_ws;
    size_t off = 0;
    bf16_t* xn  = (bf16_t*)(ws + off); off += (size_t)MM * EE * sizeof(bf16_t);
    bf16_t* wb  = (bf16_t*)(ws + off); off += (size_t)NN * EE * sizeof(bf16_t);
    float*  qkv = (float*) (ws + off); off += (size_t)MM * NN * sizeof(float);
    bf16_t* qh  = (bf16_t*)(ws + off); off += (size_t)BB * HH * TT * DD * sizeof(bf16_t);
    bf16_t* kh  = (bf16_t*)(ws + off); off += (size_t)BB * HH * TT * DD * sizeof(bf16_t);
    bf16_t* vt  = (bf16_t*)(ws + off); off += (size_t)BB * HH * TT * DD * sizeof(bf16_t);

    float* xout = (float*)d_out;
    float* y    = (float*)d_out + (size_t)MM * EE;

    rmsnorm_x_kernel<<<MM, 256, 0, stream>>>(x, xn, xout);
    convert_w_kernel<<<4096, 256, 0, stream>>>(w, wb, NN * EE);
    qkv_gemm_kernel<<<dim3(MM / 128, NN / 256), 256, 0, stream>>>(xn, wb, qkv);
    rope_rms_kernel<<<MM * HH, 128, 0, stream>>>(qkv, cosp, sinp, qh, kh, vt);
    flash_attn_kernel<<<BB * HH * (TT / 64), 128, 0, stream>>>(qh, kh, vt, y);
}
